// EdgeConv_73830487818479
// MI455X (gfx1250) — matrix operand order repaired and checked by reading, not compiled, not run on hardware
//
#include <hip/hip_runtime.h>

typedef __attribute__((ext_vector_type(16))) __bf16 v16bf;
typedef __attribute__((ext_vector_type(8)))  __bf16 v8bf;
typedef __attribute__((ext_vector_type(4)))  __bf16 v4bf;
typedef __attribute__((ext_vector_type(8)))  float  v8f;
typedef __attribute__((ext_vector_type(4)))  float  v4f;

#define NPTS   32768   // N*P = 32*1024 points
#define KROWS  16      // K neighbors -> one WMMA M-tile
#define CIN    128     // 2C feature dim
#define HDIM   64
#define WPB    4       // waves per block (128 threads)
#define BLOCKS 1024
#define ITERS  (NPTS / (BLOCKS * WPB))   // 8

// A-fragment for v_wmma_f32_16x16x32_bf16 (16-bit A 16x32):
// contiguous k order matching B: lane group h holds K 16h..16h+15.
// => two contiguous 16-byte chunks per lane.
__device__ __forceinline__ v16bf load_a_frag(const __bf16* row, int kbase, int h) {
  const __bf16* p = row + kbase + h * 16;
  v8bf lo = *(const v8bf*)(p);
  v8bf hi = *(const v8bf*)(p + 8);
  return __builtin_shufflevector(lo, hi, 0, 1, 2, 3, 4, 5, 6, 7,
                                         8, 9, 10, 11, 12, 13, 14, 15);
}

__global__ __launch_bounds__(WPB * 32)
void edgeconv_mlp_kernel(const float* __restrict__ x,
                         const float* __restrict__ W1,
                         const float* __restrict__ b1,
                         const float* __restrict__ W2,
                         const float* __restrict__ b2,
                         float* __restrict__ out) {
  __shared__ __align__(32) __bf16 sW1t[HDIM * CIN];        // [n][k], k contiguous
  __shared__ __align__(32) __bf16 sW2t[HDIM * HDIM];       // [n][k]
  __shared__ __align__(32) __bf16 sFeat[WPB][KROWS * CIN]; // per-wave 16x128 bf16
  __shared__ __align__(32) __bf16 sH1[WPB][KROWS * HDIM];  // per-wave 16x64 bf16

  const int tid  = threadIdx.x;
  const int wave = tid >> 5;
  const int lane = tid & 31;
  const int m    = lane & 15;  // row within 16-row tile / column index base
  const int h    = lane >> 4;  // lane-group half

  // Stage transposed bf16 weights once per block (B-fragment friendly layout).
  // These stay hot in cache/LDS; x traffic is marked non-temporal so it
  // doesn't evict them.
  for (int idx = tid; idx < HDIM * CIN; idx += (int)blockDim.x) {
    int n = idx >> 7, k = idx & (CIN - 1);
    sW1t[idx] = (__bf16)W1[k * HDIM + n];
  }
  for (int idx = tid; idx < HDIM * HDIM; idx += (int)blockDim.x) {
    int n = idx >> 6, k = idx & (HDIM - 1);
    sW2t[idx] = (__bf16)W2[k * HDIM + n];
  }

  float bias1[4], bias2[4];
#pragma unroll
  for (int nt = 0; nt < 4; ++nt) {
    bias1[nt] = b1[m + nt * 16];
    bias2[nt] = b2[m + nt * 16];
  }
  __syncthreads();

  const int gwave = (int)blockIdx.x * WPB + wave;

  for (int it = 0; it < ITERS; ++it) {
    const int q = it * (BLOCKS * WPB) + gwave;  // point index in [0, 32768)

    // ---- Phase 1: feat = [xi, xj - xi], f32 -> bf16, into LDS ----
    // lane covers row m, 32 cols starting at h*32; each x element read once.
    // Non-temporal: x is 268 MB streamed once (> 192 MB L2), don't cache it.
    const float* xr = x + (size_t)q * KROWS * CIN + (size_t)m * CIN + h * 32;
    __bf16* frow = &sFeat[wave][m * CIN + h * 32];
#pragma unroll
    for (int v = 0; v < 8; ++v) {
      v4f xi = __builtin_nontemporal_load((const v4f*)(xr + v * 4));
      v4f xj = __builtin_nontemporal_load((const v4f*)(xr + 64 + v * 4));
      v4bf a = { (__bf16)xi[0], (__bf16)xi[1], (__bf16)xi[2], (__bf16)xi[3] };
      v4bf d = { (__bf16)(xj[0] - xi[0]), (__bf16)(xj[1] - xi[1]),
                 (__bf16)(xj[2] - xi[2]), (__bf16)(xj[3] - xi[3]) };
      *(v4bf*)(frow + v * 4)      = a;
      *(v4bf*)(frow + 64 + v * 4) = d;
    }

    // Prefetch next iteration's tile so its HBM fetch overlaps this
    // iteration's WMMAs (global_prefetch_b8; each lane owns two 128-B
    // aligned chunks of the next tile).
    if (it + 1 < ITERS) {
      const float* xn = xr + (size_t)(BLOCKS * WPB) * KROWS * CIN;
      __builtin_prefetch(xn, 0, 1);
      __builtin_prefetch(xn + 64, 0, 1);
    }
    __syncthreads();

    // ---- GEMM1: (16x128) x (128x64), bf16 WMMA, f32 accum ----
    v8f acc[4];
#pragma unroll
    for (int nt = 0; nt < 4; ++nt) acc[nt] = (v8f){0.f,0.f,0.f,0.f,0.f,0.f,0.f,0.f};
    const __bf16* frowA = &sFeat[wave][m * CIN];
#pragma unroll
    for (int kt = 0; kt < 4; ++kt) {
      v16bf a = load_a_frag(frowA, kt * 32, h);
#pragma unroll
      for (int nt = 0; nt < 4; ++nt) {
        v16bf b = *(const v16bf*)&sW1t[(m + nt * 16) * CIN + kt * 32 + h * 16];
        acc[nt] = __builtin_amdgcn_wmma_f32_16x16x32_bf16(
            false, a, false, b, (short)0, acc[nt], false, false);
      }
    }

    // bias + ReLU -> bf16 h1 in LDS (D layout: VGPR r holds row r + h*8)
#pragma unroll
    for (int nt = 0; nt < 4; ++nt) {
      int n = m + nt * 16;
#pragma unroll
      for (int r = 0; r < 8; ++r) {
        float val = acc[nt][r] + bias1[nt];
        val = val > 0.f ? val : 0.f;
        sH1[wave][(r + h * 8) * HDIM + n] = (__bf16)val;
      }
    }
    __syncthreads();

    // ---- GEMM2: (16x64) x (64x64) ----
    v8f acc2[4];
#pragma unroll
    for (int nt = 0; nt < 4; ++nt) acc2[nt] = (v8f){0.f,0.f,0.f,0.f,0.f,0.f,0.f,0.f};
    const __bf16* hrow = &sH1[wave][m * HDIM];
#pragma unroll
    for (int kt = 0; kt < 2; ++kt) {
      v16bf a = load_a_frag(hrow, kt * 32, h);
#pragma unroll
      for (int nt = 0; nt < 4; ++nt) {
        v16bf b = *(const v16bf*)&sW2t[(m + nt * 16) * HDIM + kt * 32 + h * 16];
        acc2[nt] = __builtin_amdgcn_wmma_f32_16x16x32_bf16(
            false, a, false, b, (short)0, acc2[nt], false, false);
      }
    }

    // bias + ReLU + mean over the 16 rows, write output (non-temporal)
#pragma unroll
    for (int nt = 0; nt < 4; ++nt) {
      int n = m + nt * 16;
      float s = 0.f;
#pragma unroll
      for (int r = 0; r < 8; ++r) {
        float val = acc2[nt][r] + bias2[nt];
        s += (val > 0.f ? val : 0.f);
      }
      // lane L and L^16 hold the same column n, complementary row halves
      s += __shfl_xor(s, 16, 32);
      if (h == 0)
        __builtin_nontemporal_store(s * 0.0625f, &out[(size_t)q * HDIM + n]);
    }
    __syncthreads();  // protect sFeat/sH1 before next iteration
  }
}

extern "C" void kernel_launch(void* const* d_in, const int* in_sizes, int n_in,
                              void* d_out, int out_size, void* d_ws, size_t ws_size,
                              hipStream_t stream) {
  (void)in_sizes; (void)n_in; (void)out_size; (void)d_ws; (void)ws_size;
  const float* x  = (const float*)d_in[0];
  const float* W1 = (const float*)d_in[1];
  const float* b1 = (const float*)d_in[2];
  const float* W2 = (const float*)d_in[3];
  const float* b2 = (const float*)d_in[4];
  edgeconv_mlp_kernel<<<BLOCKS, WPB * 32, 0, stream>>>(x, W1, b1, W2, b2,
                                                       (float*)d_out);
}
